// Mutag_GCN_DiffPool8_52750788329771
// MI455X (gfx1250) — matrix operand order repaired and checked by reading, not compiled, not run on hardware
//
#include <hip/hip_runtime.h>
#include <hip/hip_bf16.h>
#include <math.h>

typedef __attribute__((ext_vector_type(16))) _Float16 v16h;
typedef __attribute__((ext_vector_type(8)))  float    v8f;
typedef __attribute__((ext_vector_type(4)))  _Float16 v4h;
typedef __attribute__((ext_vector_type(4)))  unsigned int u32x4;
typedef __attribute__((ext_vector_type(4)))  float    f32x4;

#define BB   256
#define NN   512
#define HH   30
#define HPAD 32

// ---------------------------------------------------------------------------
// 1) dis[b][i] = clip(rowsum(adj with diag forced to 1), 1)^-0.5   (wave/row)
// ---------------------------------------------------------------------------
__global__ __launch_bounds__(256) void rowsum_kernel(const float* __restrict__ adj,
                                                     float* __restrict__ dis) {
  int wid  = (blockIdx.x * 256 + threadIdx.x) >> 5;   // B*512 waves
  int lane = threadIdx.x & 31;
  int b = wid >> 9, i = wid & 511;
  const float* row = adj + ((size_t)(b * NN + i)) * NN;
  float s = 0.f;
  for (int j = lane; j < NN; j += 32) s += row[j];
  for (int off = 16; off; off >>= 1) s += __shfl_xor(s, off, 32);
  if (lane == 0) {
    s = s - row[i] + 1.f;            // force diagonal to 1
    s = fmaxf(s, 1.f);
    dis[b * NN + i] = rsqrtf(s);
  }
}

// ---------------------------------------------------------------------------
// 2) A_norm[b][i][j] = f16( dis_i * a_ij * dis_j ),  a_ii = 1     (4 elt/thr)
// ---------------------------------------------------------------------------
__global__ __launch_bounds__(256) void normstore_kernel(const float* __restrict__ adj,
                                                        const float* __restrict__ dis,
                                                        _Float16* __restrict__ An) {
  size_t idx = (size_t)blockIdx.x * 256 + threadIdx.x;   // B*512*512/4 threads
  size_t e   = idx * 4;
  int j  = (int)(e & 511);
  int bi = (int)(e >> 9);          // b*512 + i
  int i  = bi & 511;
  int b  = bi >> 9;
  f32x4 v = *(const f32x4*)(adj + e);
  float arr[4] = {v.x, v.y, v.z, v.w};
  if (i >= j && i < j + 4) arr[i - j] = 1.f;   // self-loop
  float di = dis[bi];
  const float* dj = dis + b * NN + j;
  v4h o;
  #pragma unroll
  for (int k = 0; k < 4; ++k) o[k] = (_Float16)(di * arr[k] * dj[k]);
  *(v4h*)(An + e) = o;
}

// ---------------------------------------------------------------------------
// 3) XwT[b][f][node] = f16( sum_k X[b][node][k] * W[k][f] )   (f padded to 32)
// ---------------------------------------------------------------------------
__global__ __launch_bounds__(256) void xw_kernel(const float* __restrict__ X, int fin,
                                                 int xstride, const float* __restrict__ W,
                                                 _Float16* __restrict__ XwT) {
  int idx  = blockIdx.x * 256 + threadIdx.x;   // B*512*32
  int node = idx & 511;
  int f    = (idx >> 9) & 31;
  int b    = idx >> 14;
  float acc = 0.f;
  if (f < HH) {
    const float* xr = X + (size_t)(b * NN + node) * xstride;
    for (int k = 0; k < fin; ++k) acc += xr[k] * W[k * HH + f];
  }
  XwT[(size_t)b * (HPAD * NN) + f * NN + node] = (_Float16)acc;
}

// ---------------------------------------------------------------------------
// 4) X[b] = relu?( A_norm[b] @ Xw[b] + bias )   via v_wmma_f32_16x16x32_f16
//    one wave -> one 16x32 output tile; K loop 16 x 32
// ---------------------------------------------------------------------------
__global__ __launch_bounds__(256) void gcn_gemm_kernel(const _Float16* __restrict__ An,
                                                       const _Float16* __restrict__ XwT,
                                                       const float* __restrict__ bias,
                                                       float* __restrict__ X, int relu) {
  int wave  = (blockIdx.x << 3) + (threadIdx.x >> 5);   // B*32 tiles
  int lane  = threadIdx.x & 31;
  int b     = wave >> 5;
  int m0    = (wave & 31) << 4;
  int laneM = lane & 15;
  int hi    = lane >> 4;

  const _Float16* Arow  = An  + ((size_t)(b * NN + m0 + laneM)) * NN;
  const _Float16* Bbase = XwT + (size_t)b * (HPAD * NN);
  const _Float16* bp0   = Bbase + (size_t)laneM        * NN;
  const _Float16* bp1   = Bbase + (size_t)(16 + laneM) * NN;

  v8f acc0 = {}; v8f acc1 = {};
  union F { v16h v; u32x4 u[2]; };

  for (int kt = 0; kt < 16; ++kt) {
    int k0 = kt << 5;
    F a, b0, b1;
    // A frag: lane<16 holds K k0..k0+15; lane>=16 holds k0+16..k0+31
    a.u[0] = *(const u32x4*)(Arow + k0 + hi * 16);
    a.u[1] = *(const u32x4*)(Arow + k0 + hi * 16 + 8);
    // B frag: lane<16 holds K k0..k0+15 of column laneM; lane>=16 holds k0+16..k0+31
    b0.u[0] = *(const u32x4*)(bp0 + k0 + hi * 16);
    b0.u[1] = *(const u32x4*)(bp0 + k0 + hi * 16 + 8);
    b1.u[0] = *(const u32x4*)(bp1 + k0 + hi * 16);
    b1.u[1] = *(const u32x4*)(bp1 + k0 + hi * 16 + 8);
    acc0 = __builtin_amdgcn_wmma_f32_16x16x32_f16(false, a.v, false, b0.v, (short)0, acc0, false, false);
    acc1 = __builtin_amdgcn_wmma_f32_16x16x32_f16(false, a.v, false, b1.v, (short)0, acc1, false, false);
  }

  int row0 = m0 + hi * 8;
  int c1   = 16 + laneM;
  float bv0 = bias[laneM];
  float bv1 = (c1 < HH) ? bias[c1] : 0.f;
  float* Xb = X + (size_t)(b * NN) * HPAD;
  #pragma unroll
  for (int r = 0; r < 8; ++r) {
    float v0 = acc0[r] + bv0;
    float v1 = acc1[r] + bv1;
    if (relu) { v0 = fmaxf(v0, 0.f); v1 = fmaxf(v1, 0.f); }
    Xb[(size_t)(row0 + r) * HPAD + laneM] = v0;
    Xb[(size_t)(row0 + r) * HPAD + c1]   = v1;
  }
}

// ---------------------------------------------------------------------------
// 5) Pool-0 logits: S[b][n][c] = X[b][n][:30] . pw0[:,c] + pb0[c]
// ---------------------------------------------------------------------------
__global__ __launch_bounds__(256) void pool_logits_kernel(const float* __restrict__ X,
                                                          const float* __restrict__ pw,
                                                          const float* __restrict__ pb,
                                                          float* __restrict__ S) {
  int idx  = blockIdx.x * 256 + threadIdx.x;   // B*512*8
  int cc   = idx & 7;
  int node = (idx >> 3) & 511;
  int b    = idx >> 12;
  const float* xr = X + (size_t)(b * NN + node) * HPAD;
  float acc = pb[cc];
  for (int f = 0; f < HH; ++f) acc += xr[f] * pw[f * 8 + cc];
  S[idx] = acc;
}

// 6) softmax over node dim (axis=1): one wave per (b, cluster)
__global__ __launch_bounds__(256) void softmax_nodes_kernel(float* __restrict__ S) {
  int wid  = (blockIdx.x * 256 + threadIdx.x) >> 5;   // B*8 waves
  int lane = threadIdx.x & 31;
  int b = wid >> 3, cc = wid & 7;
  float v[16];
  float m = -1e30f;
  #pragma unroll
  for (int k = 0; k < 16; ++k) {
    v[k] = S[((size_t)(b * NN + lane + 32 * k)) * 8 + cc];
    m = fmaxf(m, v[k]);
  }
  for (int off = 16; off; off >>= 1) m = fmaxf(m, __shfl_xor(m, off, 32));
  float sum = 0.f;
  #pragma unroll
  for (int k = 0; k < 16; ++k) { v[k] = expf(v[k] - m); sum += v[k]; }
  for (int off = 16; off; off >>= 1) sum += __shfl_xor(sum, off, 32);
  float inv = 1.f / sum;
  #pragma unroll
  for (int k = 0; k < 16; ++k)
    S[((size_t)(b * NN + lane + 32 * k)) * 8 + cc] = v[k] * inv;
}

// 7) softmax over cluster dim (axis=-1): one thread per (b, node)
__global__ __launch_bounds__(256) void softmax_cluster_kernel(float* __restrict__ S) {
  int idx = blockIdx.x * 256 + threadIdx.x;   // B*512
  float* r = S + (size_t)idx * 8;
  float m = -1e30f;
  #pragma unroll
  for (int c = 0; c < 8; ++c) m = fmaxf(m, r[c]);
  float sum = 0.f;
  float e[8];
  #pragma unroll
  for (int c = 0; c < 8; ++c) { e[c] = expf(r[c] - m); sum += e[c]; }
  float inv = 1.f / sum;
  #pragma unroll
  for (int c = 0; c < 8; ++c) r[c] = e[c] * inv;
}

// 8) T[b][i][c] = sum_j adj[b][i][j] * S[b][j][c]   (wave per row, fp32 adj)
__global__ __launch_bounds__(256) void adj_s_kernel(const float* __restrict__ adj,
                                                    const float* __restrict__ S,
                                                    float* __restrict__ T) {
  int wid  = (blockIdx.x * 256 + threadIdx.x) >> 5;   // B*512 waves
  int lane = threadIdx.x & 31;
  int b = wid >> 9, i = wid & 511;
  const float* row = adj + ((size_t)(b * NN + i)) * NN;
  float acc[8] = {0.f, 0.f, 0.f, 0.f, 0.f, 0.f, 0.f, 0.f};
  for (int j = lane; j < NN; j += 32) {
    float a = row[j];
    const float* sr = S + ((size_t)(b * NN + j)) * 8;
    #pragma unroll
    for (int c = 0; c < 8; ++c) acc[c] += a * sr[c];
  }
  #pragma unroll
  for (int c = 0; c < 8; ++c)
    for (int off = 16; off; off >>= 1) acc[c] += __shfl_xor(acc[c], off, 32);
  if (lane == 0) {
    float* tr = T + ((size_t)(b * NN + i)) * 8;
    #pragma unroll
    for (int c = 0; c < 8; ++c) tr[c] = acc[c];
  }
}

// 9) xp = s^T x  [8x32 padded],  ap = s^T (adj s) [8x8]   (block per batch)
__global__ __launch_bounds__(256) void xp_ap_kernel(const float* __restrict__ S,
                                                    const float* __restrict__ X,
                                                    const float* __restrict__ T,
                                                    float* __restrict__ XP,
                                                    float* __restrict__ AP) {
  int b = blockIdx.x, t = threadIdx.x;
  if (t < 240) {
    int cc = t / HH, f = t % HH;
    float acc = 0.f;
    for (int n = 0; n < NN; ++n)
      acc += S[((size_t)(b * NN + n)) * 8 + cc] * X[((size_t)(b * NN + n)) * HPAD + f];
    XP[b * 256 + cc * HPAD + f] = acc;
  } else {
    int q = t - 240; int cc = q >> 1; int f = HH + (q & 1);
    XP[b * 256 + cc * HPAD + f] = 0.f;
  }
  if (t < 64) {
    int cc = t >> 3, dd = t & 7;
    float acc = 0.f;
    for (int n = 0; n < NN; ++n)
      acc += S[((size_t)(b * NN + n)) * 8 + cc] * T[((size_t)(b * NN + n)) * 8 + dd];
    AP[b * 64 + cc * 8 + dd] = acc;
  }
}

// ---------------------------------------------------------------------------
// 10) Tail: (gcn n=8, pool->4, gcn, pool->2, gcn, pool->1, gcn, linear)
// ---------------------------------------------------------------------------
__device__ void gcn_small(float* x, const float* adjm, int n, const float* W,
                          const float* bias, float* xw, float* dis, int t) {
  if (t < n * HH) {
    int nn = t / HH, f = t % HH;
    float acc = 0.f;
    for (int k = 0; k < HH; ++k) acc += x[nn * HH + k] * W[k * HH + f];
    xw[t] = acc;
  }
  if (t < n) {
    float s = 0.f;
    for (int j = 0; j < n; ++j) s += (j == t) ? 1.f : adjm[t * 8 + j];
    s = fmaxf(s, 1.f);
    dis[t] = rsqrtf(s);
  }
  __syncthreads();
  if (t < n * HH) {
    int nn = t / HH, f = t % HH;
    float acc = 0.f;
    for (int j = 0; j < n; ++j) {
      float a = (j == nn) ? 1.f : adjm[nn * 8 + j];
      acc += dis[nn] * a * dis[j] * xw[j * HH + f];
    }
    acc += bias[f];
    acc = fmaxf(acc, 0.f);   // every tail GCN is followed by relu
    x[t] = acc;
  }
  __syncthreads();
}

__global__ __launch_bounds__(256) void tail_kernel(const float* __restrict__ XP,
    const float* __restrict__ AP,
    const float* cw4, const float* cb4, const float* cw5, const float* cb5,
    const float* cw6, const float* cb6, const float* cw7, const float* cb7,
    const float* pw1, const float* pb1, const float* pw2, const float* pb2,
    const float* pw3, const float* pb3, const float* lw, const float* lb,
    float* __restrict__ out) {
  __shared__ float x[240], xw[240], adjm[64], an[64], s[64], dis[8];
  int b = blockIdx.x, t = threadIdx.x;
  if (t < 240) { int nn = t / HH, f = t % HH; x[t] = XP[b * 256 + nn * HPAD + f]; }
  if (t < 64) adjm[t] = AP[b * 64 + t];
  __syncthreads();

  const float* cws[3] = {cw4, cw5, cw6};
  const float* cbs[3] = {cb4, cb5, cb6};
  const float* pws[3] = {pw1, pw2, pw3};
  const float* pbs[3] = {pb1, pb2, pb3};

  int n = 8;
  for (int st = 0; st < 3; ++st) {
    int c = n >> 1;
    gcn_small(x, adjm, n, cws[st], cbs[st], xw, dis, t);
    // pool logits (store with fixed stride 8)
    if (t < n * c) {
      int nn = t / c, cc = t % c;
      float acc = pbs[st][cc];
      for (int f = 0; f < HH; ++f) acc += x[nn * HH + f] * pws[st][f * c + cc];
      s[nn * 8 + cc] = acc;
    }
    __syncthreads();
    // softmax over nodes (axis=1), per cluster column
    if (t < c) {
      float m = -1e30f;
      for (int nn = 0; nn < n; ++nn) m = fmaxf(m, s[nn * 8 + t]);
      float sum = 0.f;
      for (int nn = 0; nn < n; ++nn) { float e = expf(s[nn * 8 + t] - m); s[nn * 8 + t] = e; sum += e; }
      float inv = 1.f / sum;
      for (int nn = 0; nn < n; ++nn) s[nn * 8 + t] *= inv;
    }
    __syncthreads();
    // softmax over clusters (axis=-1), per node
    if (t < n) {
      float m = -1e30f;
      for (int cc = 0; cc < c; ++cc) m = fmaxf(m, s[t * 8 + cc]);
      float sum = 0.f;
      for (int cc = 0; cc < c; ++cc) { float e = expf(s[t * 8 + cc] - m); s[t * 8 + cc] = e; sum += e; }
      float inv = 1.f / sum;
      for (int cc = 0; cc < c; ++cc) s[t * 8 + cc] *= inv;
    }
    __syncthreads();
    // xp = s^T x ; ap = s^T adj s
    if (t < c * HH) {
      int cc = t / HH, f = t % HH;
      float acc = 0.f;
      for (int nn = 0; nn < n; ++nn) acc += s[nn * 8 + cc] * x[nn * HH + f];
      xw[t] = acc;
    }
    if (t < c * c) {
      int cc = t / c, dd = t % c;
      float acc = 0.f;
      for (int i = 0; i < n; ++i)
        for (int j = 0; j < n; ++j)
          acc += s[i * 8 + cc] * adjm[i * 8 + j] * s[j * 8 + dd];
      an[t] = acc;
    }
    __syncthreads();
    if (t < c * HH) x[t] = xw[t];
    if (t < c * c) adjm[(t / c) * 8 + (t % c)] = an[t];
    __syncthreads();
    n = c;
  }
  // final GCN at n=1 (relu), then linear head
  gcn_small(x, adjm, 1, cw7, cb7, xw, dis, t);
  if (t < 2) {
    float acc = lb[t];
    for (int f = 0; f < HH; ++f) acc += x[f] * lw[f * 2 + t];
    out[b * 2 + t] = acc;
  }
}

// ---------------------------------------------------------------------------
extern "C" void kernel_launch(void* const* d_in, const int* in_sizes, int n_in,
                              void* d_out, int out_size, void* d_ws, size_t ws_size,
                              hipStream_t stream) {
  const float* x_in = (const float*)d_in[0];
  const float* adj  = (const float*)d_in[1];
  const float* cw[8]; const float* cb[8];
  for (int i = 0; i < 8; ++i) { cw[i] = (const float*)d_in[4 + 2 * i]; cb[i] = (const float*)d_in[5 + 2 * i]; }
  const float* pw[4]; const float* pb[4];
  for (int i = 0; i < 4; ++i) { pw[i] = (const float*)d_in[20 + 2 * i]; pb[i] = (const float*)d_in[21 + 2 * i]; }
  const float* lw = (const float*)d_in[28];
  const float* lb = (const float*)d_in[29];
  float* out = (float*)d_out;

  char* ws = (char*)d_ws;
  size_t off = 0;
  _Float16* An  = (_Float16*)(ws + off); off += (size_t)BB * NN * NN * 2;      // 134 MB f16 (fits L2)
  float*    dis = (float*)(ws + off);    off += (size_t)BB * NN * 4;
  _Float16* XwT = (_Float16*)(ws + off); off += (size_t)BB * HPAD * NN * 2;
  float*    X   = (float*)(ws + off);    off += (size_t)BB * NN * HPAD * 4;
  float*    S   = (float*)(ws + off);    off += (size_t)BB * NN * 8 * 4;
  float*    T   = (float*)(ws + off);    off += (size_t)BB * NN * 8 * 4;
  float*    XP  = (float*)(ws + off);    off += (size_t)BB * 8 * HPAD * 4;
  float*    AP  = (float*)(ws + off);    off += (size_t)BB * 64 * 4;
  (void)ws_size; (void)in_sizes; (void)n_in; (void)out_size;

  // normalize adjacency once -> f16 A_norm
  rowsum_kernel<<<(BB * NN) / 8, 256, 0, stream>>>(adj, dis);
  normstore_kernel<<<(BB * NN * NN / 4) / 256, 256, 0, stream>>>(adj, dis, An);

  // GCN layers 0..3 at N=512 (WMMA); relu on 0..2
  xw_kernel<<<(BB * NN * HPAD) / 256, 256, 0, stream>>>(x_in, 14, 14, cw[0], XwT);
  gcn_gemm_kernel<<<(BB * 32) / 8, 256, 0, stream>>>(An, XwT, cb[0], X, 1);
  for (int l = 1; l < 4; ++l) {
    xw_kernel<<<(BB * NN * HPAD) / 256, 256, 0, stream>>>(X, HH, HPAD, cw[l], XwT);
    gcn_gemm_kernel<<<(BB * 32) / 8, 256, 0, stream>>>(An, XwT, cb[l], X, (l < 3) ? 1 : 0);
  }

  // DiffPool 0 (512 -> 8)
  pool_logits_kernel<<<(BB * NN * 8) / 256, 256, 0, stream>>>(X, pw[0], pb[0], S);
  softmax_nodes_kernel<<<(BB * 8 * 32) / 256, 256, 0, stream>>>(S);
  softmax_cluster_kernel<<<(BB * NN) / 256, 256, 0, stream>>>(S);
  adj_s_kernel<<<(BB * NN) / 8, 256, 0, stream>>>(adj, S, T);
  xp_ap_kernel<<<BB, 256, 0, stream>>>(S, X, T, XP, AP);

  // Tail: gcn4 + pool1..3 + gcn5..7 + linear
  tail_kernel<<<BB, 256, 0, stream>>>(XP, AP,
      cw[4], cb[4], cw[5], cb[5], cw[6], cb[6], cw[7], cb[7],
      pw[1], pb[1], pw[2], pb[2], pw[3], pb[3], lw, lb, out);
}
